// tmgp_additive_17282948399643
// MI455X (gfx1250) — matrix operand order repaired and checked by reading, not compiled, not run on hardware
//
#include <hip/hip_runtime.h>

typedef __attribute__((ext_vector_type(16))) _Float16 v16h;
typedef __attribute__((ext_vector_type(8)))  float    v8f;
typedef __attribute__((ext_vector_type(4)))  unsigned int u32x4;
typedef __attribute__((ext_vector_type(4)))  int       i32x4;
typedef __attribute__((ext_vector_type(8)))  int       i32x8;

#define N_SAMP  1024
#define D_FEAT  64
#define M_PTS   1023
#define KPAD    1024

#define BM 128          // block tile rows  (8 waves x 16)
#define BN 128          // block tile cols  (8 x 16-wide wmma subtiles per wave)
#define BK 32           // wmma K per step
#define BSTRIDE 40      // LDS halves per B row (32 + 8 pad) == TDM pad_interval/amount below

// ---------------- kernel 1: per-feature min / 1/(max-min) ----------------
__global__ __launch_bounds__(256)
void tmgp_minmax_kernel(const float* __restrict__ x, float* __restrict__ stats) {
  __shared__ float smin[256];
  __shared__ float smax[256];
  const int f = blockIdx.x;
  const int t = threadIdx.x;
  float mn =  3.3e38f, mx = -3.3e38f;
  for (int s = t; s < N_SAMP; s += 256) {
    float v = x[s * D_FEAT + f];
    mn = fminf(mn, v);
    mx = fmaxf(mx, v);
  }
  smin[t] = mn; smax[t] = mx;
  __syncthreads();
  for (int off = 128; off > 0; off >>= 1) {
    if (t < off) {
      smin[t] = fminf(smin[t], smin[t + off]);
      smax[t] = fmaxf(smax[t], smax[t + off]);
    }
    __syncthreads();
  }
  if (t == 0) {
    stats[f]      = smin[0];
    stats[64 + f] = 1.0f / (smax[0] - smin[0]);
  }
}

// ------ kernel 2: pack chol_inv -> f16 transposed+padded Bt[N=1024][K=1024],
//        and design points -> interleaved {e^p, e^-p} table ------
__global__ __launch_bounds__(256)
void tmgp_pack_kernel(const float* __restrict__ chol, const float* __restrict__ pts,
                      _Float16* __restrict__ Bt, float* __restrict__ ptab) {
  const int idx  = blockIdx.x * 256 + threadIdx.x;     // 0 .. 1024*1024-1
  const int nIdx = idx >> 10;
  const int kIdx = idx & (KPAD - 1);
  float v = 0.0f;
  if (nIdx < M_PTS && kIdx < M_PTS)
    v = chol[(size_t)kIdx * M_PTS + nIdx];             // Bt[n][k] = chol[k][n]
  Bt[idx] = (_Float16)v;
  if (idx < KPAD) {
    const float p = (idx < M_PTS) ? pts[idx] : 0.0f;   // pad irrelevant: B row 1023 is zero
    ptab[2 * idx + 0] = __expf(p);                     // e^{+p}
    ptab[2 * idx + 1] = __expf(-p);                    // e^{-p}
  }
}

// ---------------- TDM: async load of one B tile (BN x BK halves) into LDS ----------------
// 2D tile: tile_dim0 = 32 elems (2B), tile_dim1 = 128 rows, row stride 1024 elems.
// LDS padding via D#: pad_interval=3 (16 DWORDs = 64B = one row), pad_amount=3 (4 DWORDs = 8 halves)
// -> packed LDS row pitch = 40 halves = BSTRIDE.
__device__ static inline void tdm_load_tile(const _Float16* gptr, unsigned lds_off) {
  const unsigned long long ga = (unsigned long long)gptr;
  u32x4 g0;
  g0.x = 1u;                                              // count=1 (valid), user mode
  g0.y = lds_off;                                         // lds_addr (bytes)
  g0.z = (unsigned)(ga & 0xFFFFFFFFu);                    // global_addr[31:0]
  g0.w = (unsigned)((ga >> 32) & 0x01FFFFFFu) | (2u << 30); // global_addr[56:32] | type=2
  i32x8 g1;
  g1[0] = (1 << 16) | (1 << 20) | (3 << 22) | (3 << 25);  // data_size=2B, pad_en, pad_interval, pad_amount
  g1[1] = (int)((KPAD & 0xFFFF) << 16);                   // tensor_dim0[15:0] @ bits 63:48
  g1[2] = (int)((KPAD & 0xFFFF) << 16);                   // dim0 hi=0 | tensor_dim1[15:0] @ 95:80
  g1[3] = (int)(BK << 16);                                // dim1 hi=0 | tile_dim0=32
  g1[4] = BN;                                             // tile_dim1=128, tile_dim2=0
  g1[5] = KPAD;                                           // tensor_dim0_stride lo32
  g1[6] = 0;                                              // stride0 hi16 | dim1_stride[15:0]
  g1[7] = 16;                                             // dim1_stride[47:16] -> stride = 2^20 (unused: tile_dim2=0)
  const i32x4 z4 = {0, 0, 0, 0};
#if __clang_major__ >= 23
  const i32x8 z8 = {0, 0, 0, 0, 0, 0, 0, 0};
  __builtin_amdgcn_tensor_load_to_lds(g0, g1, z4, z4, z8, 0);
#else
  __builtin_amdgcn_tensor_load_to_lds(g0, g1, z4, z4, 0);
#endif
}

// ---------------- kernel 3: fused k_star generation + WMMA GEMM ----------------
__global__ __launch_bounds__(256)
void tmgp_gemm_kernel(const float* __restrict__ x, const float* __restrict__ stats,
                      const float* __restrict__ ptab, const _Float16* __restrict__ Bt,
                      float* __restrict__ out) {
  __shared__ _Float16 bs[2][BN][BSTRIDE];   // 20480 B : double-buffered B tiles, [n][k]
  __shared__ float    pt[2 * KPAD];         //  8192 B : interleaved {e^p, e^-p}

  const int tid  = threadIdx.x;
  const int lane = tid & 31;
  const int wave = tid >> 5;
  const int half = lane >> 4;
  const int l16  = lane & 15;

  const int cBase = blockIdx.x * BN;
  const int rBase = blockIdx.y * BM;

  // stage exp tables once
  for (int i = tid; i < 2 * KPAD; i += 256) pt[i] = ptab[i];

  // this lane's matrix row (A-layout: M = lane%16 for both halves)
  const int row = rBase + wave * 16 + l16;
  const int f   = row >> 10;                 // xf = transpose(xs) flattened: row = f*1024 + s
  const int s   = row & (N_SAMP - 1);
  const float xs   = (x[s * D_FEAT + f] - stats[f]) * stats[64 + f];
  const float exs  = __expf(xs);             // only 2 TRANS ops per thread total:
  const float exsn = __expf(-xs);            // WMMA owns the TRANS slot in the loop

  v8f acc[8] = {};

  // triangular chol_inv: B[k][j]==0 for k>j -> only K < cBase+BN contributes
  int kMax = cBase + BN;
  if (kMax > KPAD) kMax = KPAD;
  const int kSteps = kMax / BK;

  const _Float16* gTile = Bt + (size_t)cBase * KPAD;
  const unsigned ldsOff[2] = { (unsigned)(unsigned long long)&bs[0][0][0],
                               (unsigned)(unsigned long long)&bs[1][0][0] };

  if (wave == 0) tdm_load_tile(gTile, ldsOff[0]);   // prologue: tile 0 -> buf 0

  for (int ks = 0; ks < kSteps; ++ks) {
    const int kBase = ks * BK;
    const int buf   = ks & 1;

    if (wave == 0) __builtin_amdgcn_s_wait_tensorcnt(0);  // tile ks landed
    __syncthreads();                                      // visible to all; prev buf free
    if (wave == 0 && ks + 1 < kSteps)
      tdm_load_tile(gTile + (size_t)(kBase + BK), ldsOff[1 - buf]); // overlap next tile

    // ---- A fragment via separable Laplace kernel: a = min(e^{-xs}e^{p}, e^{xs}e^{-p})
    union { float4 v[4]; float f[16]; } q0, q1;
    const float* pg0 = &pt[2 * (kBase + half * 8)];        // elems i=0..7
    const float* pg1 = &pt[2 * (kBase + 16 + half * 8)];   // elems i=8..15
#pragma unroll
    for (int j = 0; j < 4; ++j) {
      q0.v[j] = *(const float4*)(pg0 + 4 * j);
      q1.v[j] = *(const float4*)(pg1 + 4 * j);
    }
    v16h a;
#pragma unroll
    for (int i = 0; i < 8; ++i) {
      a[i]     = (_Float16)fminf(exsn * q0.f[2 * i], exs * q0.f[2 * i + 1]);
      a[i + 8] = (_Float16)fminf(exsn * q1.f[2 * i], exs * q1.f[2 * i + 1]);
    }

    // ---- issue ALL 16 ds_load_b128 first, fence, then 8 back-to-back WMMAs.
    // sched_barrier(0) keeps the loads from sinking between the WMMAs, so the
    // eight per-WMMA "s_wait_dscnt 0" stalls collapse into one.
    union BF { v16h h; u32x4 u[2]; } bf[8];
#pragma unroll
    for (int t8 = 0; t8 < 8; ++t8) {
      const _Float16* bptr = &bs[buf][t8 * 16 + l16][half * 8];
      bf[t8].u[0] = *(const u32x4*)(bptr);
      bf[t8].u[1] = *(const u32x4*)(bptr + 16);
    }
    __builtin_amdgcn_sched_barrier(0);
#pragma unroll
    for (int t8 = 0; t8 < 8; ++t8)
      acc[t8] = __builtin_amdgcn_wmma_f32_16x16x32_f16(
          false, a, false, bf[t8].h, (short)0, acc[t8], false, false);

    asm volatile("s_wait_dscnt 0x0" ::: "memory");  // LDS reads done before next TDM overwrite
  }

  // epilogue: C/D layout -> VGPR v: row = rBase+wave*16 + v + half*8, col = subtile + l16
#pragma unroll
  for (int t8 = 0; t8 < 8; ++t8) {
    const int col = cBase + t8 * 16 + l16;
    if (col < M_PTS) {
#pragma unroll
      for (int v = 0; v < 8; ++v) {
        const int r = rBase + wave * 16 + v + half * 8;
        out[(size_t)r * M_PTS + col] = acc[t8][v];
      }
    }
  }
}

// ---------------- host-side launch ----------------
extern "C" void kernel_launch(void* const* d_in, const int* in_sizes, int n_in,
                              void* d_out, int out_size, void* d_ws, size_t ws_size,
                              hipStream_t stream) {
  (void)in_sizes; (void)n_in; (void)out_size; (void)ws_size;
  const float* x    = (const float*)d_in[0];   // [1024, 64]
  const float* pts  = (const float*)d_in[1];   // [1023, 1]
  const float* chol = (const float*)d_in[2];   // [1023, 1023]
  float* out = (float*)d_out;                  // [65536 * 1023] flat

  char* ws = (char*)d_ws;
  float*    stats = (float*)ws;                // 128 f32 @ 0
  float*    ptab  = (float*)(ws + 512);        // 2048 f32 @ 512
  _Float16* Bt    = (_Float16*)(ws + 8704);    // 1024*1024 f16 (2 MB), 256B aligned

  hipLaunchKernelGGL(tmgp_minmax_kernel, dim3(64),     dim3(256), 0, stream, x, stats);
  hipLaunchKernelGGL(tmgp_pack_kernel,   dim3(4096),   dim3(256), 0, stream, chol, pts, Bt, ptab);
  hipLaunchKernelGGL(tmgp_gemm_kernel,   dim3(8, 512), dim3(256), 0, stream,
                     x, stats, ptab, Bt, out);
}
